// MoShead_41145786696383
// MI455X (gfx1250) — matrix operand order repaired and checked by reading, not compiled, not run on hardware
//
#include <hip/hip_runtime.h>
#include <hip/hip_bf16.h>
#include <stdint.h>
#include <stddef.h>

#define NTOK  32000
#define NINPE 512
#define NHID  1024
#define NEXP  10
#define NROW  1024
#define NLAT  (NEXP*NINPE)   // 5120

typedef __bf16 bf16;
typedef __attribute__((ext_vector_type(8)))  bf16  bf16x8;
typedef __attribute__((ext_vector_type(16))) bf16  bf16x16;
typedef __attribute__((ext_vector_type(8)))  float f32x8;

union FragAB { bf16x16 v; bf16x8 h[2]; };

// ---------------- CDNA5 async global->LDS copy (ASYNCcnt path) ------------
// LDS offset = low 32 bits of generic ptr.
__device__ inline void async_cp16(uint32_t lds_off, const void* gptr) {
  asm volatile("global_load_async_to_lds_b128 %0, %1, off"
               :: "v"(lds_off), "v"((uint64_t)(uintptr_t)gptr)
               : "memory");
}
__device__ inline void wait_async0() {
  asm volatile("s_wait_asynccnt 0x0" ::: "memory");
}
// Pin: force acc to be materialized here, so WMMAs schedule BEFORE the
// following volatile s_wait_asynccnt (volatile asms stay ordered).
__device__ inline void pin_acc(f32x8& a) {
  asm volatile("" : "+v"(a));
}

// ---- WMMA fragment loads per CDNA5 ISA layouts (wave32) -------
// A: 16x32 bf16 tile in LDS, row-major, row stride 40 bf16 (80B, 16B aligned)
__device__ inline FragAB load_a(const bf16* __restrict__ base, int lane) {
  FragAB f;
  const int row  = lane & 15;
  const int half = lane >> 4;
  const bf16* p = base + row * 40;
  f.h[0] = *(const bf16x8*)(p + half * 8);        // V0..3 : K 0..7  / 8..15
  f.h[1] = *(const bf16x8*)(p + 16 + half * 8);   // V4..7 : K16..23 / 24..31
  return f;
}
// B: 32x16 bf16, stored in LDS as [N=16 rows][K=32] (i.e. B^T, K-contiguous)
__device__ inline FragAB load_b(const bf16* __restrict__ base, int lane) {
  FragAB f;
  const int col   = lane & 15;
  const int halfk = lane >> 4;
  const bf16* p = base + col * 40 + halfk * 8;    // lanes0-15: K0..7,16..23, lanes16-31: K8..15,24..31
  f.h[0] = *(const bf16x8*)(p);
  f.h[1] = *(const bf16x8*)(p + 16);
  return f;
}

// --------------------------- f32 -> bf16 convert --------------------------
__global__ void cvt_f32_bf16(const float* __restrict__ s, bf16* __restrict__ d, int n) {
  for (int i = blockIdx.x * blockDim.x + threadIdx.x; i < n; i += gridDim.x * blockDim.x)
    d[i] = (bf16)s[i];
}

__global__ void zero_f32(float* __restrict__ p, int n) {
  int i = blockIdx.x * blockDim.x + threadIdx.x;
  if (i < n) p[i] = 0.0f;
}

// ------------------- latent = tanh(out @ Wlat^T + b) ----------------------
// grid (NLAT/128, NROW/16), 256 threads (8 waves); wave w -> cols w*16..+16
__global__ __launch_bounds__(256) void latent_gemm(
    const bf16* __restrict__ outb,   // [1024][1024]
    const bf16* __restrict__ wlat,   // [5120][1024] (k-contiguous)
    const float* __restrict__ b_lat, // [5120]
    bf16* __restrict__ latent)       // [1024][5120]
{
  __shared__ __align__(16) bf16 At[2][16][40];
  __shared__ __align__(16) bf16 Bt[2][128][40];
  const int tid = threadIdx.x, lane = tid & 31, wave = tid >> 5;
  const int cBase = blockIdx.x * 128, nBase = blockIdx.y * 16;
  f32x8 acc = {0.f,0.f,0.f,0.f,0.f,0.f,0.f,0.f};

  auto stage = [&](int buf, int kBase) {
    if (tid < 64) {                            // A: 16 rows x 32 k
      int r = tid >> 2, c = tid & 3;
      async_cp16((uint32_t)(uintptr_t)&At[buf][r][c * 8],
                 &outb[(size_t)(nBase + r) * NHID + kBase + c * 8]);
    }
#pragma unroll
    for (int it = 0; it < 2; ++it) {           // B: 128 cols x 32 k
      int i = tid + it * 256;
      int row = i >> 2, c = i & 3;
      async_cp16((uint32_t)(uintptr_t)&Bt[buf][row][c * 8],
                 &wlat[(size_t)(cBase + row) * NHID + kBase + c * 8]);
    }
  };

  stage(0, 0);
  wait_async0();
  __syncthreads();

  const int KS = NHID / 32;
  for (int ks = 0; ks < KS; ++ks) {
    const int cur = ks & 1;
    if (ks + 1 < KS) stage(cur ^ 1, (ks + 1) * 32);   // overlap with WMMA
    FragAB a = load_a(&At[cur][0][0], lane);
    FragAB b = load_b(&Bt[cur][wave * 16][0], lane);
    acc = __builtin_amdgcn_wmma_f32_16x16x32_bf16(false, a.v, false, b.v, (short)0, acc, false, false);
    pin_acc(acc);                                      // keep WMMA before the wait
    if (ks + 1 < KS) { wait_async0(); __syncthreads(); }
  }

  const int col = lane & 15, half = lane >> 4;
  const int c = cBase + wave * 16 + col;
  const float bl = b_lat[c];
#pragma unroll
  for (int j = 0; j < 8; ++j) {
    int n = nBase + j + half * 8;
    latent[(size_t)n * NLAT + c] = (bf16)tanhf(acc[j] + bl);
  }
}

// ----------------------- mixture weights (pis) ----------------------------
__device__ inline uint64_t splitmix64(uint64_t z) {
  z += 0x9E3779B97F4A7C15ull;
  z = (z ^ (z >> 30)) * 0xBF58476D1CE4E5B9ull;
  z = (z ^ (z >> 27)) * 0x94D049BB133111EBull;
  return z ^ (z >> 31);
}
__device__ inline float u01(uint64_t& s) {
  s = splitmix64(s);
  return ((float)(uint32_t)(s >> 40) + 0.5f) * (1.0f / 16777216.0f);
}
__device__ float gamma_sample(float alpha, uint64_t& s) {
  float boost = 1.0f;
  if (alpha < 1.0f) { boost = __powf(u01(s), 1.0f / alpha); alpha += 1.0f; }
  const float d = alpha - (1.0f / 3.0f);
  const float c = __frsqrt_rn(9.0f * d);
  for (int it = 0; it < 24; ++it) {   // Marsaglia-Tsang, deterministic stream
    float u1 = u01(s), u2 = u01(s);
    float x = __fsqrt_rn(-2.0f * __logf(u1)) * __cosf(6.28318530718f * u2);
    float v = 1.0f + c * x;
    v = v * v * v;
    if (v <= 0.0f) continue;
    float u = u01(s);
    if (__logf(u) < 0.5f * x * x + d - d * v + d * __logf(v)) return boost * d * v;
  }
  return boost * d;
}
__device__ inline float softplus_f(float x) {
  return (x > 20.0f) ? x : log1pf(__expf(x));
}

__global__ __launch_bounds__(32) void pis_kernel(
    const float* __restrict__ outp,  // [1024][1024] f32
    const float* __restrict__ wred,  // [10][1024]
    const float* __restrict__ bred,  // [10]
    float* __restrict__ pis)         // [1024][10]
{
  const int n = blockIdx.x, lane = threadIdx.x;
  float x[32];
  for (int i = 0; i < 32; ++i) x[i] = outp[(size_t)n * NHID + lane + i * 32];
  float av[NEXP];
  for (int e = 0; e < NEXP; ++e) {
    float s = 0.0f;
    for (int i = 0; i < 32; ++i) s = fmaf(x[i], wred[(size_t)e * NHID + lane + i * 32], s);
    for (int m = 1; m < 32; m <<= 1) s += __shfl_xor(s, m, 32);
    av[e] = s;
  }
  if (lane == 0) {
    float a[NEXP], bsum = 0.0f;
    for (int e = 0; e < NEXP; ++e) { a[e] = softplus_f(av[e] + bred[e]) + 1e-8f; bsum += a[e]; }
    float csum = 0.0f, rp = 1.0f;
    for (int e = 0; e < NEXP; ++e) {
      csum += a[e];
      float dd = fmaxf(fabsf(bsum - csum), 1e-8f);
      uint64_t seed = splitmix64(0x42ull ^ ((uint64_t)(n * NEXP + e) << 20));
      float g1 = gamma_sample(a[e], seed);
      float g2 = gamma_sample(dd, seed);
      float smp = g1 / (g1 + g2 + 1e-20f);
      pis[n * NEXP + e] = rp * smp;
      rp *= (1.0f - smp);
    }
  }
}

// --------- decoder GEMM passes: MODE 0 = sum-exp(Z), MODE 1 = combine -----
// grid (NTOK/256, NROW/16), 256 threads (8 waves); wave w -> tokens w*32..+32
// expert-sliced A (cols e*512+k of latent) => no cross-n row interleave
template <int MODE>
__global__ __launch_bounds__(256) void moe_gemm(
    const bf16*  __restrict__ latent, // [1024][5120]
    const bf16*  __restrict__ wdec,   // [32000][512] (k-contiguous)
    const float* __restrict__ b_dec,  // [32000]
    const float* __restrict__ pis,    // [1024*10]
    float*       __restrict__ Z,      // [10240], r = n*10+e
    float*       __restrict__ out)    // [1024][32000]
{
  __shared__ __align__(16) bf16 At[2][NEXP][16][40];   // 25.6 KB
  __shared__ __align__(16) bf16 Bt[2][256][40];        // 41.0 KB
  const int tid = threadIdx.x, lane = tid & 31, wave = tid >> 5;
  const int tBase = blockIdx.x * 256, nBase = blockIdx.y * 16;

  f32x8 acc[NEXP][2];
#pragma unroll
  for (int e = 0; e < NEXP; ++e)
#pragma unroll
    for (int q = 0; q < 2; ++q)
      acc[e][q] = f32x8{0.f,0.f,0.f,0.f,0.f,0.f,0.f,0.f};

  auto stage = [&](int buf, int kBase) {
#pragma unroll
    for (int it = 0; it < 4; ++it) {           // B: 256 tokens x 32 k (1024 chunks)
      int i = tid + it * 256;
      int row = i >> 2, c = i & 3;
      async_cp16((uint32_t)(uintptr_t)&Bt[buf][row][c * 8],
                 &wdec[(size_t)(tBase + row) * NINPE + kBase + c * 8]);
    }
#pragma unroll
    for (int it = 0; it < 2; ++it) {           // A: 10e x 16n x 32k (640 chunks)
      int i = tid + it * 256;
      int e = i >> 6, r = (i >> 2) & 15, c = i & 3;
      async_cp16((uint32_t)(uintptr_t)&At[buf][e][r][c * 8],
                 &latent[(size_t)(nBase + r) * NLAT + e * NINPE + kBase + c * 8]);
    }
    if (tid < 128) {
      int i = tid + 512;
      int e = i >> 6, r = (i >> 2) & 15, c = i & 3;
      async_cp16((uint32_t)(uintptr_t)&At[buf][e][r][c * 8],
                 &latent[(size_t)(nBase + r) * NLAT + e * NINPE + kBase + c * 8]);
    }
  };

  stage(0, 0);
  wait_async0();
  __syncthreads();

  const int KS = NINPE / 32;
  for (int ks = 0; ks < KS; ++ks) {
    const int cur = ks & 1;
    if (ks + 1 < KS) stage(cur ^ 1, (ks + 1) * 32);   // async copies overlap WMMAs
    FragAB b0 = load_b(&Bt[cur][wave * 32][0], lane);
    FragAB b1 = load_b(&Bt[cur][wave * 32 + 16][0], lane);
#pragma unroll
    for (int e = 0; e < NEXP; ++e) {
      FragAB a = load_a(&At[cur][e][0][0], lane);
      acc[e][0] = __builtin_amdgcn_wmma_f32_16x16x32_bf16(false, a.v, false, b0.v, (short)0,
                                                          acc[e][0], false, false);
      acc[e][1] = __builtin_amdgcn_wmma_f32_16x16x32_bf16(false, a.v, false, b1.v, (short)0,
                                                          acc[e][1], false, false);
      pin_acc(acc[e][0]);                               // keep WMMAs before the wait
      pin_acc(acc[e][1]);
    }
    if (ks + 1 < KS) { wait_async0(); __syncthreads(); }
  }

  const int col = lane & 15, half = lane >> 4;
  const int t0 = tBase + wave * 32 + col;
  const int t1 = t0 + 16;
  const float bd0 = b_dec[t0];
  const float bd1 = b_dec[t1];

  if (MODE == 0) {
#pragma unroll
    for (int e = 0; e < NEXP; ++e) {
#pragma unroll
      for (int j = 0; j < 8; ++j) {
        float s = __expf(acc[e][0][j] + bd0) + __expf(acc[e][1][j] + bd1);
        for (int m = 1; m < 16; m <<= 1) s += __shfl_xor(s, m, 32); // 16-lane row sum
        if (col == 0) {
          int n = nBase + j + half * 8;
          atomicAdd(&Z[n * NEXP + e], s);
        }
      }
    }
  } else {
    float o0[8] = {0.f,0.f,0.f,0.f,0.f,0.f,0.f,0.f};
    float o1[8] = {0.f,0.f,0.f,0.f,0.f,0.f,0.f,0.f};
#pragma unroll
    for (int e = 0; e < NEXP; ++e) {
#pragma unroll
      for (int j = 0; j < 8; ++j) {
        int n = nBase + j + half * 8;
        float sc = pis[n * NEXP + e] / Z[n * NEXP + e];
        o0[j] += __expf(acc[e][0][j] + bd0) * sc;
        o1[j] += __expf(acc[e][1][j] + bd1) * sc;
      }
    }
#pragma unroll
    for (int j = 0; j < 8; ++j) {
      int n = nBase + j + half * 8;
      out[(size_t)n * NTOK + t0] = o0[j];
      out[(size_t)n * NTOK + t1] = o1[j];
    }
  }
}

// ------------------------------- launcher ---------------------------------
extern "C" void kernel_launch(void* const* d_in, const int* in_sizes, int n_in,
                              void* d_out, int out_size, void* d_ws, size_t ws_size,
                              hipStream_t stream) {
  const float* output  = (const float*)d_in[0];  // [1024,1024]
  const float* Wlat    = (const float*)d_in[1];  // [5120,1024]
  const float* b_lat   = (const float*)d_in[2];  // [5120]
  const float* Wdec    = (const float*)d_in[3];  // [32000,512]
  const float* b_dec   = (const float*)d_in[4];  // [32000]
  const float* Wred    = (const float*)d_in[5];  // [10,1024]
  const float* b_red   = (const float*)d_in[6];  // [10]
  float* out = (float*)d_out;

  char* ws = (char*)d_ws;
  size_t off = 0;
  bf16* latent_bf = (bf16*)(ws + off); off += (size_t)NROW * NLAT * sizeof(bf16);   // 10.5 MB
  bf16* wdec_bf   = (bf16*)(ws + off); off += (size_t)NTOK * NINPE * sizeof(bf16);  // 32.8 MB
  bf16* wlat_bf   = (bf16*)(ws + off); off += (size_t)NLAT * NHID * sizeof(bf16);   // 10.5 MB
  bf16* out_bf    = (bf16*)(ws + off); off += (size_t)NROW * NHID * sizeof(bf16);   //  2.1 MB
  float* Zbuf     = (float*)(ws + off); off += (size_t)NROW * NEXP * sizeof(float);
  float* pis      = (float*)(ws + off); off += (size_t)NROW * NEXP * sizeof(float);

  // 1) convert operands to bf16
  cvt_f32_bf16<<<2048, 256, 0, stream>>>(Wdec,   wdec_bf, NTOK * NINPE);
  cvt_f32_bf16<<<2048, 256, 0, stream>>>(Wlat,   wlat_bf, NLAT * NHID);
  cvt_f32_bf16<<<1024, 256, 0, stream>>>(output, out_bf,  NROW * NHID);

  // 2) latent = tanh(output @ Wlat^T + b_lat)  [bf16 out]
  latent_gemm<<<dim3(NLAT / 128, NROW / 16), 256, 0, stream>>>(out_bf, wlat_bf, b_lat, latent_bf);

  // 3) stick-breaking mixture weights
  pis_kernel<<<NROW, 32, 0, stream>>>(output, Wred, b_red, pis);

  // 4) Z = rowwise sum of exp(logit)  (zero, then atomic accumulate)
  zero_f32<<<(NROW * NEXP + 255) / 256, 256, 0, stream>>>(Zbuf, NROW * NEXP);
  moe_gemm<0><<<dim3(NTOK / 256, NROW / 16), 256, 0, stream>>>(latent_bf, wdec_bf, b_dec, pis, Zbuf, out);

  // 5) out[n,t] = sum_e pis[n,e] * exp(logit)/Z
  moe_gemm<1><<<dim3(NTOK / 256, NROW / 16), 256, 0, stream>>>(latent_bf, wdec_bf, b_dec, pis, Zbuf, out);
}
